// ADRNN_4063039062461
// MI455X (gfx1250) — matrix operand order repaired and checked by reading, not compiled, not run on hardware
//
#include <hip/hip_runtime.h>
#include <hip/hip_bf16.h>
#include <math.h>

// ---------------------------------------------------------------------------
// ADRNN (dual 2-layer LSTM) for MI455X / gfx1250, wave32 + WMMA bf16.
//   gates = [x_pad | h_prev] @ [Wih ; Whh]^T  (single fused bf16 WMMA GEMM)
//   fused epilogue: bias + sigmoid/tanh + c update (f32) + h stores (bf16)
// LSTM stage: block = 8 waves sharing one hidden N-tile; B tiles staged
// through LDS (double-buffered chunks, templated so the pipeline is fully
// unrolled straight-line code), B tiles double-buffered in registers so
// WMMAs never wait on a just-issued ds_load.
// ---------------------------------------------------------------------------

typedef __attribute__((ext_vector_type(16))) __bf16 v16bf;
typedef __attribute__((ext_vector_type(8)))  __bf16 v8bf;
typedef __attribute__((ext_vector_type(8)))  float  v8f;

#define HSZ   512
#define BATCH 128
#define TT    512
#define RSZ   47

// ------------------------- workspace layout (bytes) ------------------------
static const size_t OFF_WR0 = 0;                                   // KT=18 NT=128
static const size_t OFF_WR1 = OFF_WR0 + (size_t)18 * 128 * 1024;   // KT=32
static const size_t OFF_WT0 = OFF_WR1 + (size_t)32 * 128 * 1024;   // KT=20
static const size_t OFF_WT1 = OFF_WT0 + (size_t)20 * 128 * 1024;   // KT=32
static const size_t OFF_WRO = OFF_WT1 + (size_t)32 * 128 * 1024;   // KT=16 NT=3
static const size_t OFF_WTO = OFF_WRO + (size_t)16 * 3   * 1024;   // KT=16 NT=1
static const size_t OFF_ACT = OFF_WTO + (size_t)16 * 1   * 1024;
static const size_t SZ_AR0 = (size_t)BATCH * 576  * 2;   // [x_pad64 | hr0]
static const size_t SZ_AR1 = (size_t)BATCH * 1024 * 2;   // [hr0 | hr1]
static const size_t SZ_AT0 = (size_t)BATCH * 640  * 2;   // [x49|rout47|pad|ht0]
static const size_t SZ_AT1 = (size_t)BATCH * 1024 * 2;   // [ht0 | ht1]
static const size_t SZ_H   = (size_t)BATCH * HSZ  * 2;
static const size_t SZ_C   = (size_t)BATCH * HSZ  * 4;
static const size_t OFF_AR0 = OFF_ACT;
static const size_t OFF_AR1 = OFF_AR0 + 2 * SZ_AR0;
static const size_t OFF_AT0 = OFF_AR1 + 2 * SZ_AR1;
static const size_t OFF_AT1 = OFF_AT0 + 2 * SZ_AT0;
static const size_t OFF_HR1 = OFF_AT1 + 2 * SZ_AT1;
static const size_t OFF_HT1 = OFF_HR1 + SZ_H;
static const size_t OFF_CR0 = OFF_HT1 + SZ_H;
static const size_t OFF_CR1 = OFF_CR0 + SZ_C;
static const size_t OFF_CT0 = OFF_CR1 + SZ_C;
static const size_t OFF_CT1 = OFF_CT0 + SZ_C;
static const size_t OFF_END = OFF_CT1 + SZ_C;
static const size_t ACT_BYTES = OFF_END - OFF_ACT;

// ---------------------------------------------------------------------------
__global__ void adrnn_zero_kernel(uint4* __restrict__ p, unsigned n16) {
    unsigned i = blockIdx.x * blockDim.x + threadIdx.x;
    if (i < n16) p[i] = make_uint4(0u, 0u, 0u, 0u);
}

// ---------------------------------------------------------------------------
// Pack fused [W0(K0real, pad to K0pad) ; W1(K1)]^T into WMMA B tiles (bf16).
// B tile (32x16): lane L, elem j holds B[k = kt*32 + table(L>>4, j)][n = nt*16 + (L&15)] = W[n][k].
__global__ void adrnn_pack_kernel(__bf16* __restrict__ dst,
                                  const float* __restrict__ W0, int K0real, int K0pad,
                                  const float* __restrict__ W1, int K1,
                                  int Nreal, int KT, int NT) {
    int idx = blockIdx.x * blockDim.x + threadIdx.x;
    int total = NT * KT * 32;
    if (idx >= total) return;
    int nt   = idx / (KT * 32);
    int rem  = idx - nt * (KT * 32);
    int kt   = rem >> 5;
    int lane = rem & 31;
    int n    = nt * 16 + (lane & 15);
    __bf16* out = dst + ((size_t)(nt * KT + kt) * 32 + lane) * 16;
    #pragma unroll 4
    for (int j = 0; j < 16; ++j) {
        int k = kt * 32 + (j < 8 ? 8 * (lane >> 4) + j : 16 + 8 * (lane >> 4) + (j - 8));
        float v = 0.f;
        if (n < Nreal) {
            if (k < K0pad) {
                if (k < K0real) v = W0[(size_t)n * K0real + k];
            } else if (K1 > 0) {
                v = W1[(size_t)n * K1 + (k - K0pad)];
            }
        }
        out[j] = (__bf16)v;
    }
}

// ---------------------------------------------------------------------------
__global__ void adrnn_xstage_kernel(const float* __restrict__ x_r,
                                    const float* __restrict__ x_t, int t,
                                    __bf16* __restrict__ ar0,
                                    __bf16* __restrict__ at0) {
    int b = blockIdx.x;
    int col = threadIdx.x;     // 0..63
    float v = 0.f;
    if (col < 47)      v = x_r[(size_t)b * TT * 47 + (size_t)t * 47 + col];
    else if (col < 49) v = x_t[(size_t)b * TT * 2  + (size_t)t * 2 + (col - 47)];
    __bf16 hb = (__bf16)v;
    ar0[(size_t)b * 576 + col] = hb;
    at0[(size_t)b * 640 + col] = hb;
}

// ---------------------------------------------------------------------------
// A-tile loader: 16-bit A 16x32 WMMA layout from row-major bf16 [rows][lda].
__device__ __forceinline__ v16bf adrnn_load_a(const __bf16* __restrict__ A,
                                              int lda, int mT, int kt, int lane) {
    int row  = mT * 16 + (lane & 15);
    int half = lane >> 4;
    const __bf16* p = A + (size_t)row * lda + kt * 32 + half * 8;
    v8bf lo = *(const v8bf*)p;
    v8bf hi = *(const v8bf*)(p + 16);
    v16bf a;
    #pragma unroll
    for (int i = 0; i < 8; ++i) { a[i] = lo[i]; a[i + 8] = hi[i]; }
    return a;
}

__device__ __forceinline__ float adrnn_sigmoid(float x) {
    return 1.0f / (1.0f + __expf(-x));
}

// ---------------------------------------------------------------------------
// Fused LSTM cell stage, LDS-pipelined. CKS must divide KT (no tail guards ->
// straight-line unrolled pipeline). Grid: 32 blocks (ntH) x 256 threads
// (8 waves, wave = mT). Per chunk: A-tile loads first, then next-chunk global
// fetch (so A waits don't drag the fetch), then LDS->reg double-buffered
// WMMA groups, then reg->LDS stash (where the fetch's loadcnt wait lands).
template <int KT, int CKS>
__global__ __launch_bounds__(256) void adrnn_lstm_kernel(
        const __bf16* __restrict__ A, int lda,
        const __bf16* __restrict__ Bpk,
        const float* __restrict__ bias,
        float* __restrict__ Cst,
        __bf16* __restrict__ d0, int ld0, int off0,
        __bf16* __restrict__ d1, int ld1, int off1) {
    constexpr int NC = KT / CKS;                 // chunks (exact)
    __shared__ __bf16 Blds[2][CKS * 4 * 512];    // 2 x (4 gates x CKS KB)

    int tid  = threadIdx.x;
    int lane = tid & 31;
    int mT   = tid >> 5;          // batch tile 0..7 (one per wave)
    int ntH  = blockIdx.x;        // hidden tile 0..31

    uint4 freg[CKS];
    // fetch chunk c's B tiles (4 gates x CKS tiles = CKS*4 KB) into registers
    auto fetch = [&](int c) {
        int kbase = c * CKS;
        #pragma unroll
        for (int i = 0; i < CKS; ++i) {
            int u  = tid + i * 256;              // 16-byte unit
            int tl = u >> 6;                     // tile slot = g*CKS + kc
            int g  = tl / CKS;
            int kc = tl % CKS;
            const __bf16* src = Bpk
                + ((size_t)(ntH + g * 32) * KT + (kbase + kc)) * 512
                + (size_t)(u & 63) * 8;
            freg[i] = *(const uint4*)src;
        }
    };
    auto stash = [&](int c) {
        int buf = c & 1;
        #pragma unroll
        for (int i = 0; i < CKS; ++i) {
            int u  = tid + i * 256;
            int tl = u >> 6;
            *(uint4*)&Blds[buf][(size_t)tl * 512 + (size_t)(u & 63) * 8] = freg[i];
        }
    };

    v8f ai = {}, af = {}, ag = {}, ao = {};
    v16bf breg[2][4];

    fetch(0);
    stash(0);
    for (int c = 0; c < NC; ++c) {
        __syncthreads();                         // LDS[c&1] ready for reads
        int kbase = c * CKS;

        // 1) A tiles of this chunk (issued BEFORE the next-chunk fetch so
        //    in-order LOADcnt waits on A don't also wait on the fetch)
        v16bf areg[CKS];
        #pragma unroll
        for (int kc = 0; kc < CKS; ++kc)
            areg[kc] = adrnn_load_a(A, lda, mT, kbase + kc, lane);

        // 2) next chunk's global->reg fetch, completes behind the WMMAs
        if (c + 1 < NC) fetch(c + 1);

        // 3) LDS->reg double-buffered WMMA pipeline
        const __bf16* lbase = &Blds[c & 1][lane * 16];
        #pragma unroll
        for (int g = 0; g < 4; ++g)
            breg[0][g] = *(const v16bf*)(lbase + g * (CKS * 512));
        #pragma unroll
        for (int kc = 0; kc < CKS; ++kc) {
            if (kc + 1 < CKS) {
                const __bf16* lp = lbase + (kc + 1) * 512;
                #pragma unroll
                for (int g = 0; g < 4; ++g)
                    breg[(kc + 1) & 1][g] = *(const v16bf*)(lp + g * (CKS * 512));
            }
            ai = __builtin_amdgcn_wmma_f32_16x16x32_bf16(false, areg[kc], false, breg[kc & 1][0], (short)0, ai, false, false);
            af = __builtin_amdgcn_wmma_f32_16x16x32_bf16(false, areg[kc], false, breg[kc & 1][1], (short)0, af, false, false);
            ag = __builtin_amdgcn_wmma_f32_16x16x32_bf16(false, areg[kc], false, breg[kc & 1][2], (short)0, ag, false, false);
            ao = __builtin_amdgcn_wmma_f32_16x16x32_bf16(false, areg[kc], false, breg[kc & 1][3], (short)0, ao, false, false);
        }

        // 4) commit next chunk to the other LDS buffer
        if (c + 1 < NC) stash(c + 1);
    }

    // epilogue: element v -> M = mT*16 + v + 8*(lane>>4), N = lane&15
    int half = lane >> 4;
    int nh   = ntH * 16 + (lane & 15);
    float bI = bias[nh], bF = bias[512 + nh], bG = bias[1024 + nh], bO = bias[1536 + nh];
    #pragma unroll
    for (int v = 0; v < 8; ++v) {
        int m = mT * 16 + v + 8 * half;
        float gi = adrnn_sigmoid(ai[v] + bI);
        float gf = adrnn_sigmoid(af[v] + bF);
        float gg = tanhf(ag[v] + bG);
        float go = adrnn_sigmoid(ao[v] + bO);
        size_t ci = (size_t)m * HSZ + nh;
        float cc = gf * Cst[ci] + gi * gg;
        Cst[ci] = cc;
        float h = go * tanhf(cc);
        __bf16 hb = (__bf16)h;
        d0[(size_t)m * ld0 + off0 + nh] = hb;
        if (d1) d1[(size_t)m * ld1 + off1 + nh] = hb;
    }
}

// ---------------------------------------------------------------------------
// Output linear: out = h @ W^T + b (f32, strided), optional bf16 side copy.
__global__ __launch_bounds__(32) void adrnn_linear_kernel(
        const __bf16* __restrict__ A,
        const __bf16* __restrict__ Bpk, int KT,
        const float* __restrict__ bias, int Nreal,
        float* __restrict__ out, int outRowStride,
        __bf16* __restrict__ dBf, int ldBf, int offBf) {
    int lane = threadIdx.x & 31;
    int mT = blockIdx.x & 7;
    int nt = blockIdx.x >> 3;
    v8f acc = {};
    const __bf16* Bp = Bpk + ((size_t)nt * KT) * 512 + lane * 16;
    for (int kt = 0; kt < KT; ++kt) {
        v16bf a = adrnn_load_a(A, HSZ, mT, kt, lane);
        v16bf b = *(const v16bf*)(Bp + (size_t)kt * 512);
        acc = __builtin_amdgcn_wmma_f32_16x16x32_bf16(false, a, false, b, (short)0, acc, false, false);
    }
    int half = lane >> 4;
    int n = nt * 16 + (lane & 15);
    if (n < Nreal) {
        float bb = bias[n];
        #pragma unroll
        for (int v = 0; v < 8; ++v) {
            int m = mT * 16 + v + 8 * half;
            float val = acc[v] + bb;
            out[(size_t)m * outRowStride + n] = val;
            if (dBf) dBf[(size_t)m * ldBf + offBf + n] = (__bf16)val;
        }
    }
}

// ---------------------------------------------------------------------------
extern "C" void kernel_launch(void* const* d_in, const int* in_sizes, int n_in,
                              void* d_out, int out_size, void* d_ws, size_t ws_size,
                              hipStream_t stream) {
    const float* x_r    = (const float*)d_in[0];
    const float* x_t    = (const float*)d_in[1];
    const float* r_Wih0 = (const float*)d_in[2];
    const float* r_Whh0 = (const float*)d_in[3];
    const float* r_b0   = (const float*)d_in[4];
    const float* r_Wih1 = (const float*)d_in[5];
    const float* r_Whh1 = (const float*)d_in[6];
    const float* r_b1   = (const float*)d_in[7];
    const float* t_Wih0 = (const float*)d_in[8];
    const float* t_Whh0 = (const float*)d_in[9];
    const float* t_b0   = (const float*)d_in[10];
    const float* t_Wih1 = (const float*)d_in[11];
    const float* t_Whh1 = (const float*)d_in[12];
    const float* t_b1   = (const float*)d_in[13];
    const float* Wr     = (const float*)d_in[14];
    const float* br     = (const float*)d_in[15];
    const float* Wt     = (const float*)d_in[16];
    const float* bt     = (const float*)d_in[17];

    char* ws = (char*)d_ws;
    __bf16* WR0 = (__bf16*)(ws + OFF_WR0);
    __bf16* WR1 = (__bf16*)(ws + OFF_WR1);
    __bf16* WT0 = (__bf16*)(ws + OFF_WT0);
    __bf16* WT1 = (__bf16*)(ws + OFF_WT1);
    __bf16* WRO = (__bf16*)(ws + OFF_WRO);
    __bf16* WTO = (__bf16*)(ws + OFF_WTO);
    __bf16* AR0[2] = { (__bf16*)(ws + OFF_AR0), (__bf16*)(ws + OFF_AR0 + SZ_AR0) };
    __bf16* AR1[2] = { (__bf16*)(ws + OFF_AR1), (__bf16*)(ws + OFF_AR1 + SZ_AR1) };
    __bf16* AT0[2] = { (__bf16*)(ws + OFF_AT0), (__bf16*)(ws + OFF_AT0 + SZ_AT0) };
    __bf16* AT1[2] = { (__bf16*)(ws + OFF_AT1), (__bf16*)(ws + OFF_AT1 + SZ_AT1) };
    __bf16* HR1 = (__bf16*)(ws + OFF_HR1);
    __bf16* HT1 = (__bf16*)(ws + OFF_HT1);
    float* CR0 = (float*)(ws + OFF_CR0);
    float* CR1 = (float*)(ws + OFF_CR1);
    float* CT0 = (float*)(ws + OFF_CT0);
    float* CT1 = (float*)(ws + OFF_CT1);

    unsigned n16 = (unsigned)(ACT_BYTES / 16);
    adrnn_zero_kernel<<<(n16 + 255) / 256, 256, 0, stream>>>((uint4*)(ws + OFF_ACT), n16);

    auto pack = [&](__bf16* dst, const float* W0, int K0r, int K0p,
                    const float* W1, int K1, int Nreal, int KT, int NT) {
        int total = NT * KT * 32;
        adrnn_pack_kernel<<<(total + 255) / 256, 256, 0, stream>>>(
            dst, W0, K0r, K0p, W1, K1, Nreal, KT, NT);
    };
    pack(WR0, r_Wih0, 49, 64,  r_Whh0, 512, 2048, 18, 128);   // r cell 0 (K=576)
    pack(WR1, r_Wih1, 512, 512, r_Whh1, 512, 2048, 32, 128);  // r cell 1 (K=1024)
    pack(WT0, t_Wih0, 96, 128, t_Whh0, 512, 2048, 20, 128);   // t cell 0 (K=640)
    pack(WT1, t_Wih1, 512, 512, t_Whh1, 512, 2048, 32, 128);  // t cell 1 (K=1024)
    pack(WRO, Wr, 512, 512, nullptr, 0, 47, 16, 3);           // r_out linear
    pack(WTO, Wt, 512, 512, nullptr, 0, 2, 16, 1);            // t_out linear

    float* out_r = (float*)d_out;                              // [B,T,47]
    float* out_t = out_r + (size_t)BATCH * TT * RSZ;           // [B,T,2]

    for (int t = 0; t < TT; ++t) {
        int p = t & 1, q = p ^ 1;

        adrnn_xstage_kernel<<<BATCH, 64, 0, stream>>>(x_r, x_t, t, AR0[p], AT0[p]);

        // r cell 0: A = [x|hr0(t-1)] (K=576), hr0(t) -> AR1[p][:,0:512] & AR0[q][:,64:576]
        adrnn_lstm_kernel<18, 3><<<32, 256, 0, stream>>>(AR0[p], 576, WR0, r_b0, CR0,
                                                         AR1[p], 1024, 0, AR0[q], 576, 64);

        // r cell 1: A = [hr0(t)|hr1(t-1)] (K=1024), hr1(t) -> AR1[q][:,512:1024] & HR1
        adrnn_lstm_kernel<32, 4><<<32, 256, 0, stream>>>(AR1[p], 1024, WR1, r_b1, CR1,
                                                         AR1[q], 1024, 512, HR1, 512, 0);

        // r_out = hr1 @ Wr^T + br -> d_out (f32) + AT0[p][:,49:96] (bf16)
        adrnn_linear_kernel<<<8 * 3, 32, 0, stream>>>(HR1, WRO, 16, br, 47,
                                                      out_r + (size_t)t * RSZ, TT * RSZ,
                                                      AT0[p], 640, 49);

        // t cell 0: A = [x|r_out|pad|ht0(t-1)] (K=640), ht0(t) -> AT1[p][:,0:512] & AT0[q][:,128:640]
        adrnn_lstm_kernel<20, 4><<<32, 256, 0, stream>>>(AT0[p], 640, WT0, t_b0, CT0,
                                                         AT1[p], 1024, 0, AT0[q], 640, 128);

        // t cell 1: A = [ht0(t)|ht1(t-1)] (K=1024), ht1(t) -> AT1[q][:,512:1024] & HT1
        adrnn_lstm_kernel<32, 4><<<32, 256, 0, stream>>>(AT1[p], 1024, WT1, t_b1, CT1,
                                                         AT1[q], 1024, 512, HT1, 512, 0);

        // t_out = ht1 @ Wt^T + bt -> d_out (f32)
        adrnn_linear_kernel<<<8 * 1, 32, 0, stream>>>(HT1, WTO, 16, bt, 2,
                                                      out_t + (size_t)t * 2, TT * 2,
                                                      nullptr, 0, 0);
    }
}
